// CrossAttention_27925877359245
// MI455X (gfx1250) — matrix operand order repaired and checked by reading, not compiled, not run on hardware
//
#include <hip/hip_runtime.h>
#include <hip/hip_bf16.h>

// ---------------------------------------------------------------------------
// CDNA5 (gfx1250) cross-attention: bf16 WMMA + TDM async tensor loads.
// ---------------------------------------------------------------------------

typedef __attribute__((ext_vector_type(16))) __bf16       v16bf;
typedef __attribute__((ext_vector_type(8)))  float        v8f;
typedef __attribute__((ext_vector_type(8)))  unsigned int v8u;
typedef __attribute__((ext_vector_type(4)))  unsigned int v4u;
typedef __attribute__((ext_vector_type(8)))  int          v8i;
typedef __attribute__((ext_vector_type(4)))  int          v4i;

#define DIM       768
#define NHEADS    12
#define HDIM      64
#define BATCH     4
#define SEQ       1024
#define ROWS_TOT  (BATCH * SEQ)       // 4096
#define KSTEPS    (DIM / 32)          // 24

__device__ __forceinline__ unsigned short f32_to_bf16(float f) {
    unsigned int u = __builtin_bit_cast(unsigned int, f);
    unsigned int r = u + 0x7FFFu + ((u >> 16) & 1u);   // round-to-nearest-even
    return (unsigned short)(r >> 16);
}

#if __has_builtin(__builtin_amdgcn_cvt_pk_bf16_f32)
__device__ __forceinline__ unsigned int pack2_bf16(float a, float b) {
    typedef __attribute__((ext_vector_type(2))) __bf16 v2bf;
    v2bf r = __builtin_amdgcn_cvt_pk_bf16_f32(a, b);
    return __builtin_bit_cast(unsigned int, r);
}
#else
__device__ __forceinline__ unsigned int pack2_bf16(float a, float b) {
    return (unsigned int)f32_to_bf16(a) | ((unsigned int)f32_to_bf16(b) << 16);
}
#endif

// A-fragment (16x32 bf16): lane half L<16 -> K {0..15}; L>=16 -> K {16..31}.
__device__ __forceinline__ v16bf frag_split(const unsigned short* p) {
    uint4 lo = *reinterpret_cast<const uint4*>(p);
    uint4 hi = *reinterpret_cast<const uint4*>(p + 8);
    v8u u; u[0]=lo.x; u[1]=lo.y; u[2]=lo.z; u[3]=lo.w;
           u[4]=hi.x; u[5]=hi.y; u[6]=hi.z; u[7]=hi.w;
    return __builtin_bit_cast(v16bf, u);
}

// B-fragment (32x16 bf16): lane holds one output column, 16 contiguous K values.
__device__ __forceinline__ v16bf frag_contig(const unsigned short* p) {
    uint4 lo = *reinterpret_cast<const uint4*>(p);
    uint4 hi = *reinterpret_cast<const uint4*>(p + 8);
    v8u u; u[0]=lo.x; u[1]=lo.y; u[2]=lo.z; u[3]=lo.w;
           u[4]=hi.x; u[5]=hi.y; u[6]=hi.z; u[7]=hi.w;
    return __builtin_bit_cast(v16bf, u);
}

__device__ __forceinline__ v8f wmma_bf16(v16bf a, v16bf b, v8f c) {
    return __builtin_amdgcn_wmma_f32_16x16x32_bf16(
        false, a, false, b, (short)0, c, false, false);
}

// ---------------------------------------------------------------------------
// TDM: DMA a 64x64 bf16 tile (row stride `stride_elems`) from global into LDS,
// with hardware padding of 4 DWORDs after every 32 DWORDs -> LDS pitch 72
// elements (144B), matching the fragment addressing below.
// D# layout per CDNA5 ISA ch.8 (group0 128b, group1 256b).
// This toolchain exposes the 6-arg builtin:
//   (uint32x4 g0, int32x8 g1, int32x4, int32x4, int32x8, i32 cpol)
// ---------------------------------------------------------------------------
#if __has_builtin(__builtin_amdgcn_tensor_load_to_lds)
#define HAVE_TDM 1
__device__ __forceinline__ void tdm_load_tile_64x64(
    unsigned int lds_byte_addr, const unsigned short* gptr,
    unsigned int stride_elems)
{
    const unsigned long long ga = (unsigned long long)(uintptr_t)gptr;
    v4u g0;
    g0[0] = 1u;                                        // count=1 (valid), load
    g0[1] = lds_byte_addr;                             // lds_addr
    g0[2] = (unsigned int)(ga & 0xFFFFFFFFu);          // global_addr[31:0]
    g0[3] = (unsigned int)((ga >> 32) & 0x01FFFFFFu)   // global_addr[56:32]
          | (2u << 30);                                // type = 2 ("image")
    v8i g1;
    g1[0] = (int)((1u << 16)        // data_size = 2 bytes
                | (1u << 20)        // pad_enable
                | (4u << 22)        // pad_interval: every 32 DWORDs
                | (3u << 25));      // pad_amount:   4 DWORDs (16B)
    g1[1] = (int)(64u << 16);       // tensor_dim0 = 64 (low 16 bits)
    g1[2] = (int)(64u << 16);       // tensor_dim1 = 64 (low 16 bits)
    g1[3] = (int)(64u << 16);       // tile_dim0 = 64
    g1[4] = (int)(64u);             // tile_dim1 = 64, tile_dim2 = 0
    g1[5] = (int)stride_elems;      // tensor_dim0_stride[31:0]
    g1[6] = 0;
    g1[7] = 0;
    const v4i z4 = {0, 0, 0, 0};                       // groups 2/3 unused (2-D)
    const v8i z8 = {0, 0, 0, 0, 0, 0, 0, 0};
    __builtin_amdgcn_tensor_load_to_lds(g0, g1, z4, z4, z8, 0);
}
#endif

// ---------------------------------------------------------------------------
// Kernel 1: fused QKV projection.  grid = (12 heads, 32 row-blocks, 3 mats)
//   mode 0: Q = (x @ Wq^T) * SCALE -> (B,H,N,D)   mode 1: K = y@Wk^T + yw
//   mode 2: V = y @ Wv^T -> (B,H,D,M) transposed store.
// ---------------------------------------------------------------------------
#define APITCH 40   // bf16 elements per LDS row (80B, 16B aligned)

__global__ __launch_bounds__(256) void qkv_proj_kernel(
    const float* __restrict__ x,  const float* __restrict__ y,
    const float* __restrict__ yw,
    const float* __restrict__ Wq, const float* __restrict__ Wk,
    const float* __restrict__ Wv,
    unsigned short* __restrict__ Qo, unsigned short* __restrict__ Ko,
    unsigned short* __restrict__ Vo)
{
    __shared__ __align__(16) unsigned short lds_a[128 * APITCH];
    __shared__ __align__(16) unsigned short lds_b[64  * APITCH];

    const int mode = blockIdx.z;
    const int h    = blockIdx.x;
    const int m0   = blockIdx.y * 128;
    const int n0   = h * HDIM;
    const int tid  = threadIdx.x;
    const int lane = tid & 31;
    const int wv   = tid >> 5;

    const float* A = (mode == 0) ? x : y;
    const float* W = (mode == 0) ? Wq : ((mode == 1) ? Wk : Wv);

    v8f acc[4];
#pragma unroll
    for (int c = 0; c < 4; ++c)
#pragma unroll
        for (int g = 0; g < 8; ++g) acc[c][g] = 0.0f;

    const int ar = tid >> 1, ap = (tid & 1) * 16;     // A tile: 128 rows x 32
    const int br = tid >> 2, bq = (tid & 3) * 8;      // B tile: 64 rows x 32

    for (int ks = 0; ks < KSTEPS; ++ks) {
        const int k0 = ks * 32;
        {   // stage A (fp32 -> bf16, packed stores)
            const float* src = A + (size_t)(m0 + ar) * DIM + k0 + ap;
            unsigned int* d32 = reinterpret_cast<unsigned int*>(lds_a + ar * APITCH + ap);
            if (ks + 1 < KSTEPS) __builtin_prefetch(src + 32, 0, 1);
#pragma unroll
            for (int i = 0; i < 16; i += 4) {
                float4 f = *reinterpret_cast<const float4*>(src + i);
                d32[(i >> 1) + 0] = pack2_bf16(f.x, f.y);
                d32[(i >> 1) + 1] = pack2_bf16(f.z, f.w);
            }
        }
        {   // stage W rows ([n][k], contiguous in k)
            const float* src = W + (size_t)(n0 + br) * DIM + k0 + bq;
            unsigned int* d32 = reinterpret_cast<unsigned int*>(lds_b + br * APITCH + bq);
#pragma unroll
            for (int i = 0; i < 8; i += 4) {
                float4 f = *reinterpret_cast<const float4*>(src + i);
                d32[(i >> 1) + 0] = pack2_bf16(f.x, f.y);
                d32[(i >> 1) + 1] = pack2_bf16(f.z, f.w);
            }
        }
        __syncthreads();

        const unsigned short* abase =
            lds_a + (16 * wv + (lane & 15)) * APITCH + ((lane >> 4) << 4);
        v16bf afr = frag_split(abase);
#pragma unroll
        for (int c = 0; c < 4; ++c) {
            const unsigned short* bbase =
                lds_b + (16 * c + (lane & 15)) * APITCH + ((lane >> 4) << 4);
            acc[c] = wmma_bf16(afr, frag_contig(bbase), acc[c]);
        }
        __syncthreads();
    }

    const int rbase = 16 * wv + ((lane >> 4) << 3);
#pragma unroll
    for (int g = 0; g < 8; ++g) {
        const int rglob = m0 + rbase + g;
        const int bidx  = rglob >> 10;
        const int nm    = rglob & 1023;
        const float kb  = (mode == 1) ? yw[rglob] : 0.0f;
#pragma unroll
        for (int c = 0; c < 4; ++c) {
            const int d = 16 * c + (lane & 15);
            float v = acc[c][g];
            if (mode == 0) {
                v *= 0.125f;
                Qo[((size_t)(bidx * NHEADS + h) * SEQ + nm) * HDIM + d] = f32_to_bf16(v);
            } else if (mode == 1) {
                v += kb;
                Ko[((size_t)(bidx * NHEADS + h) * SEQ + nm) * HDIM + d] = f32_to_bf16(v);
            } else {
                Vo[((size_t)(bidx * NHEADS + h) * HDIM + d) * SEQ + nm] = f32_to_bf16(v);
            }
        }
    }
}

// ---------------------------------------------------------------------------
// Kernel 2: flash-style attention.  grid = (N/64, H, B), 256 threads = 8 waves.
// K/V tiles DMA'd by the Tensor Data Mover (wave 0 issues, TENSORcnt-gated).
// ---------------------------------------------------------------------------
#define TPITCH 72   // bf16 elems / row (144B); matches TDM pad 4dw per 32dw
#define SPITCH 65   // fp32 elems / row

__global__ __launch_bounds__(256) void attention_kernel(
    const unsigned short* __restrict__ Q, const unsigned short* __restrict__ K,
    const unsigned short* __restrict__ V, unsigned short* __restrict__ O)
{
    __shared__ __align__(16) unsigned short lds_k[64 * TPITCH]; // [m][d]
    __shared__ __align__(16) unsigned short lds_v[64 * TPITCH]; // [d][m]
    __shared__ __align__(16) unsigned short lds_p[64 * TPITCH]; // Q stage, then P [n][m]
    __shared__ float lds_s[64 * SPITCH];
    __shared__ float lds_c[64];

    const int b  = blockIdx.z, h = blockIdx.y, n0 = blockIdx.x * 64;
    const int tid = threadIdx.x, lane = tid & 31, wv = tid >> 5;
    const int rg  = wv & 3;
    const int cgb = (wv >> 2) * 2;

    const size_t hoff = (size_t)(b * NHEADS + h) * SEQ * HDIM;
    const unsigned short* Qh = Q + hoff;
    const unsigned short* Kh = K + hoff;
    const unsigned short* Vh = V + hoff;   // (D, M) layout

    // stage Q tile (64x64) into lds_p, pull A-fragments, hold for whole kernel
    {
        const int r = tid >> 2, dp = (tid & 3) * 16;
        const uint4* src = reinterpret_cast<const uint4*>(Qh + (size_t)(n0 + r) * HDIM + dp);
        uint4 a = src[0], c = src[1];
        *reinterpret_cast<uint4*>(lds_p + r * TPITCH + dp)     = a;
        *reinterpret_cast<uint4*>(lds_p + r * TPITCH + dp + 8) = c;
    }
    __syncthreads();
    v16bf frag_q0, frag_q1;
    {
        const unsigned short* qb =
            lds_p + (16 * rg + (lane & 15)) * TPITCH + ((lane >> 4) << 4);
        frag_q0 = frag_split(qb);
        frag_q1 = frag_split(qb + 32);
    }
    __syncthreads();

    float m_i = -INFINITY, l_i = 0.0f;
    v8f acc_o[2];
#pragma unroll
    for (int s = 0; s < 2; ++s)
#pragma unroll
        for (int g = 0; g < 8; ++g) acc_o[s][g] = 0.0f;

    for (int mt = 0; mt < SEQ / 64; ++mt) {
        const int m0 = mt * 64;

        // --- stage K [m][d] and V [d][m] tiles into LDS ---
#ifdef HAVE_TDM
        if (wv == 0) {
            tdm_load_tile_64x64((unsigned int)(uintptr_t)lds_k,
                                Kh + (size_t)m0 * HDIM, HDIM);
            tdm_load_tile_64x64((unsigned int)(uintptr_t)lds_v,
                                Vh + m0, SEQ);
            __builtin_amdgcn_s_wait_tensorcnt(0);
        }
#else
        {
            const int r = tid >> 2, dp = (tid & 3) * 16;
            const uint4* ks = reinterpret_cast<const uint4*>(Kh + (size_t)(m0 + r) * HDIM + dp);
            uint4 a = ks[0], c = ks[1];
            *reinterpret_cast<uint4*>(lds_k + r * TPITCH + dp)     = a;
            *reinterpret_cast<uint4*>(lds_k + r * TPITCH + dp + 8) = c;
            const uint4* vs = reinterpret_cast<const uint4*>(Vh + (size_t)r * SEQ + m0 + dp);
            uint4 va = vs[0], vb = vs[1];
            *reinterpret_cast<uint4*>(lds_v + r * TPITCH + dp)     = va;
            *reinterpret_cast<uint4*>(lds_v + r * TPITCH + dp + 8) = vb;
        }
#endif
        __syncthreads();

        // --- S = Q K^T ---
        v8f acc_s[2];
#pragma unroll
        for (int s = 0; s < 2; ++s) {
#pragma unroll
            for (int g = 0; g < 8; ++g) acc_s[s][g] = 0.0f;
            const int cg = cgb + s;
            const unsigned short* kb =
                lds_k + (16 * cg + (lane & 15)) * TPITCH + ((lane >> 4) << 4);
            acc_s[s] = wmma_bf16(frag_q0, frag_contig(kb),      acc_s[s]);
            acc_s[s] = wmma_bf16(frag_q1, frag_contig(kb + 32), acc_s[s]);
        }
        {
            const int rb = 16 * rg + ((lane >> 4) << 3);
#pragma unroll
            for (int s = 0; s < 2; ++s) {
                const int col = 32 * (wv >> 2) + 16 * s + (lane & 15);
#pragma unroll
                for (int g = 0; g < 8; ++g)
                    lds_s[(rb + g) * SPITCH + col] = acc_s[s][g];
            }
        }
        __syncthreads();

        // --- online softmax: one thread per row ---
        if (tid < 64) {
            const float* srow = lds_s + tid * SPITCH;
            float mx = m_i;
#pragma unroll 8
            for (int j = 0; j < 64; ++j) mx = fmaxf(mx, srow[j]);
            const float corr = __expf(m_i - mx);
            float sum = 0.0f;
            unsigned int* prow32 = reinterpret_cast<unsigned int*>(lds_p + tid * TPITCH);
#pragma unroll 8
            for (int j = 0; j < 64; j += 2) {
                const float p0 = __expf(srow[j]     - mx);
                const float p1 = __expf(srow[j + 1] - mx);
                sum += p0 + p1;
                prow32[j >> 1] = pack2_bf16(p0, p1);
            }
            l_i = l_i * corr + sum;
            m_i = mx;
            lds_c[tid] = corr;
        }
        __syncthreads();

        // --- rescale O and accumulate O += P @ V ---
        {
            const int rb = 16 * rg + ((lane >> 4) << 3);
#pragma unroll
            for (int g = 0; g < 8; ++g) {
                const float c = lds_c[rb + g];
                acc_o[0][g] *= c;
                acc_o[1][g] *= c;
            }
            const unsigned short* pb =
                lds_p + (16 * rg + (lane & 15)) * TPITCH + ((lane >> 4) << 4);
            v16bf fp0 = frag_split(pb);
            v16bf fp1 = frag_split(pb + 32);
#pragma unroll
            for (int s = 0; s < 2; ++s) {
                const int cg = cgb + s;
                const unsigned short* vb =
                    lds_v + (16 * cg + (lane & 15)) * TPITCH + ((lane >> 4) << 4);
                acc_o[s] = wmma_bf16(fp0, frag_contig(vb),      acc_o[s]);
                acc_o[s] = wmma_bf16(fp1, frag_contig(vb + 32), acc_o[s]);
            }
        }
        __syncthreads();
    }

    if (tid < 64) lds_c[tid] = (l_i > 0.0f) ? (1.0f / l_i) : 0.0f;
    __syncthreads();

    {
        const int rb = 16 * rg + ((lane >> 4) << 3);
#pragma unroll
        for (int s = 0; s < 2; ++s) {
            const int d = 32 * (wv >> 2) + 16 * s + (lane & 15);
#pragma unroll
            for (int g = 0; g < 8; ++g) {
                const int r = rb + g;
                const float o = acc_o[s][g] * lds_c[r];
                O[((size_t)(b * SEQ + n0 + r)) * DIM + h * HDIM + d] = f32_to_bf16(o);
            }
        }
    }
}

// ---------------------------------------------------------------------------
// Kernel 3: output projection  out = O @ Wp^T + bp  (fp32 out).
// ---------------------------------------------------------------------------
__global__ __launch_bounds__(256) void out_proj_kernel(
    const unsigned short* __restrict__ Obf, const float* __restrict__ Wp,
    const float* __restrict__ bp, float* __restrict__ out)
{
    __shared__ __align__(16) unsigned short lds_a[128 * APITCH];
    __shared__ __align__(16) unsigned short lds_b[64  * APITCH];

    const int m0 = blockIdx.y * 128;
    const int n0 = blockIdx.x * 64;
    const int tid = threadIdx.x, lane = tid & 31, wv = tid >> 5;

    v8f acc[4];
#pragma unroll
    for (int c = 0; c < 4; ++c)
#pragma unroll
        for (int g = 0; g < 8; ++g) acc[c][g] = 0.0f;

    const int ar = tid >> 1, ap = (tid & 1) * 16;
    const int br = tid >> 2, bq = (tid & 3) * 8;

    for (int ks = 0; ks < KSTEPS; ++ks) {
        const int k0 = ks * 32;
        {   // A: bf16 copy
            const uint4* src = reinterpret_cast<const uint4*>(
                Obf + (size_t)(m0 + ar) * DIM + k0 + ap);
            if (ks + 1 < KSTEPS) __builtin_prefetch(src + 4, 0, 1);
            uint4 a = src[0], c = src[1];
            *reinterpret_cast<uint4*>(lds_a + ar * APITCH + ap)     = a;
            *reinterpret_cast<uint4*>(lds_a + ar * APITCH + ap + 8) = c;
        }
        {   // B: fp32 -> bf16
            const float* src = Wp + (size_t)(n0 + br) * DIM + k0 + bq;
            unsigned int* d32 = reinterpret_cast<unsigned int*>(lds_b + br * APITCH + bq);
#pragma unroll
            for (int i = 0; i < 8; i += 4) {
                float4 f = *reinterpret_cast<const float4*>(src + i);
                d32[(i >> 1) + 0] = pack2_bf16(f.x, f.y);
                d32[(i >> 1) + 1] = pack2_bf16(f.z, f.w);
            }
        }
        __syncthreads();

        const unsigned short* abase =
            lds_a + (16 * wv + (lane & 15)) * APITCH + ((lane >> 4) << 4);
        v16bf afr = frag_split(abase);
#pragma unroll
        for (int c = 0; c < 4; ++c) {
            const unsigned short* bbase =
                lds_b + (16 * c + (lane & 15)) * APITCH + ((lane >> 4) << 4);
            acc[c] = wmma_bf16(afr, frag_contig(bbase), acc[c]);
        }
        __syncthreads();
    }

    const int rbase = m0 + 16 * wv + ((lane >> 4) << 3);
#pragma unroll
    for (int c = 0; c < 4; ++c) {
        const int col = n0 + 16 * c + (lane & 15);
        const float bias = bp[col];
#pragma unroll
        for (int g = 0; g < 8; ++g)
            out[(size_t)(rbase + g) * DIM + col] = acc[c][g] + bias;
    }
}

// ---------------------------------------------------------------------------
extern "C" void kernel_launch(void* const* d_in, const int* in_sizes, int n_in,
                              void* d_out, int out_size, void* d_ws, size_t ws_size,
                              hipStream_t stream) {
    const float* x  = (const float*)d_in[0];
    const float* y  = (const float*)d_in[1];
    const float* yw = (const float*)d_in[2];
    const float* Wq = (const float*)d_in[3];
    const float* Wk = (const float*)d_in[4];
    const float* Wv = (const float*)d_in[5];
    const float* Wp = (const float*)d_in[6];
    const float* bp = (const float*)d_in[7];
    float* out = (float*)d_out;

    const size_t perTensor = (size_t)BATCH * NHEADS * SEQ * HDIM; // bf16 elements
    unsigned short* Qws = (unsigned short*)d_ws;
    unsigned short* Kws = Qws + perTensor;
    unsigned short* Vws = Kws + perTensor;
    unsigned short* Ows = Vws + perTensor;

    qkv_proj_kernel<<<dim3(NHEADS, ROWS_TOT / 128, 3), 256, 0, stream>>>(
        x, y, yw, Wq, Wk, Wv, Qws, Kws, Vws);
    attention_kernel<<<dim3(SEQ / 64, NHEADS, BATCH), 256, 0, stream>>>(
        Qws, Kws, Vws, Ows);
    out_proj_kernel<<<dim3(DIM / 64, ROWS_TOT / 128), 256, 0, stream>>>(
        Ows, Wp, bp, out);
}
